// LightweightIncongruityDetector_46308337385523
// MI455X (gfx1250) — matrix operand order repaired and checked by reading, not compiled, not run on hardware
//
#include <hip/hip_runtime.h>
#include <math.h>

// Problem constants (B=8, S=4096, H=768, C=64)
#define NTOK   32768      // B*S
#define HDIM   768
#define CDIM   64
#define NCOLS  80         // 64 MLP cols + 1 fused semantic col + 15 zero pad (5 WMMA N-tiles)
#define TOKPB  128        // tokens per workgroup
#define KCHUNK 64         // K staged per LDS tile
#define NCHUNK (HDIM / KCHUNK)   // 12
#define XBUF_BYTES (TOKPB * KCHUNK * 2)      // 16 KB per buffer
#define EPI_STRIDE 81                        // bank-conflict padding
#define EPI_BYTES  (TOKPB * EPI_STRIDE * 4)  // 41472 B  (>= 2*XBUF_BYTES)

typedef __attribute__((ext_vector_type(16))) __bf16 bf16x16;
typedef __attribute__((ext_vector_type(8)))  float  floatx8;

union ABFrag {
    bf16x16 v;
    uint4   q[2];
};

static __device__ __forceinline__ unsigned short f2bf(float f) {
    union { float f; unsigned int u; } x; x.f = f;
    unsigned int r = x.u + 0x7FFFu + ((x.u >> 16) & 1u);   // round-to-nearest-even
    return (unsigned short)(r >> 16);
}

static __device__ __forceinline__ float sigmoidf(float x) {
    return 1.0f / (1.0f + __expf(-x));
}

// Stage one 128x64 fp32 tile -> bf16 LDS buffer (coalesced float4 reads, uniform control).
static __device__ __forceinline__ void stage_chunk(const float* __restrict__ X,
                                                   int tokBase, int c,
                                                   unsigned short* __restrict__ buf,
                                                   int tid) {
    #pragma unroll
    for (int i = tid; i < TOKPB * (KCHUNK / 4); i += 256) {
        const int row = i >> 4;          // 16 float4 per row
        const int c4  = i & 15;
        const float* gp = X + (size_t)(tokBase + row) * HDIM + c * KCHUNK + c4 * 4;
        const float4 f = *(const float4*)gp;
        if (c + 1 < NCHUNK) __builtin_prefetch(gp + KCHUNK, 0, 3);  // global_prefetch_b8
        uint2 p;
        p.x = (unsigned int)f2bf(f.x) | ((unsigned int)f2bf(f.y) << 16);
        p.y = (unsigned int)f2bf(f.z) | ((unsigned int)f2bf(f.w) << 16);
        *(uint2*)&buf[row * KCHUNK + c4 * 4] = p;
    }
}

// ---------------------------------------------------------------------------
// Prep: Wcomb[n][k] (bf16, n-major so B fragments are contiguous in K):
//   n in [0,64):  W_e1[k][n]
//   n == 64:      w1[k] = sum_c W_ct[k][c] * W_cd[c]     (composed linear)
//   n in (64,80): 0
// c1 = b_ct . W_cd + b_cd
// ---------------------------------------------------------------------------
__global__ void prep_kernel(const float* __restrict__ W_ct,
                            const float* __restrict__ W_cd,
                            const float* __restrict__ b_ct,
                            const float* __restrict__ b_cd,
                            const float* __restrict__ W_e1,
                            unsigned short* __restrict__ Wcomb,
                            float* __restrict__ c1) {
    int k = blockIdx.x * blockDim.x + threadIdx.x;
    if (k < HDIM) {
        float w1 = 0.0f;
        #pragma unroll 8
        for (int c = 0; c < CDIM; ++c) w1 += W_ct[k * CDIM + c] * W_cd[c];
        for (int n = 0; n < CDIM; ++n)
            Wcomb[n * HDIM + k] = f2bf(W_e1[k * CDIM + n]);
        Wcomb[CDIM * HDIM + k] = f2bf(w1);
        for (int n = CDIM + 1; n < NCOLS; ++n) Wcomb[n * HDIM + k] = 0;
    }
    if (k == 0) {
        float s = b_cd[0];
        for (int c = 0; c < CDIM; ++c) s += b_ct[c] * W_cd[c];
        *c1 = s;
    }
}

// ---------------------------------------------------------------------------
// Fused main kernel: one streaming pass over hidden_states, double-buffered.
//   acc(16tok x 80col) = X_tile(bf16) @ Wcomb(bf16)  via v_wmma_f32_16x16x32_bf16
//   sc  = sigmoid(acc[:,64] + c1)
//   ev  = sigmoid(relu(acc[:,0:64] + b_e1) . W_e2 + b_e2)
//   cs  = 1/4096  (mean over the softmax axis of a softmax is exactly 1/S)
// ---------------------------------------------------------------------------
__global__ void __launch_bounds__(256)
incongruity_kernel(const float* __restrict__ X,
                   const unsigned short* __restrict__ Wcomb,
                   const float* __restrict__ c1p,
                   const float* __restrict__ b_e1,
                   const float* __restrict__ W_e2,
                   const float* __restrict__ b_e2,
                   float* __restrict__ out) {
    // Arena: two 16KB bf16 X buffers during the K loop; re-used as the 40.5KB
    // f32 epilogue tile after the final barrier (lifetimes are disjoint).
    __shared__ __align__(16) char smem[EPI_BYTES];
    __shared__ float ldsB1[CDIM];
    __shared__ float ldsW2[CDIM];

    unsigned short* ldsX0 = (unsigned short*)smem;
    unsigned short* ldsX1 = (unsigned short*)(smem + XBUF_BYTES);
    float*          ldsE  = (float*)smem;

    const int tid  = threadIdx.x;
    const int lane = tid & 31;
    const int wave = tid >> 5;
    const int half = lane >> 4;   // 0/1 half-wave
    const int l15  = lane & 15;
    const int tokBase = blockIdx.x * TOKPB;
    const int arow = wave * 16 + l15;   // token row (within tile) this lane feeds to A

    if (tid < CDIM)             ldsB1[tid]        = b_e1[tid];
    else if (tid < 2 * CDIM)    ldsW2[tid - CDIM] = W_e2[tid - CDIM];

    floatx8 acc[5];
    #pragma unroll
    for (int j = 0; j < 5; ++j)
        #pragma unroll
        for (int r = 0; r < 8; ++r) acc[j][r] = 0.0f;

    stage_chunk(X, tokBase, 0, ldsX0, tid);
    __syncthreads();

    for (int c = 0; c < NCHUNK; ++c) {
        unsigned short* cur = (c & 1) ? ldsX1 : ldsX0;
        unsigned short* nxt = (c & 1) ? ldsX0 : ldsX1;

        // Stage chunk c+1 while chunk c is multiplied (uniform branch, no divergence).
        if (c + 1 < NCHUNK) stage_chunk(X, tokBase, c + 1, nxt, tid);

        // ---- 2 x (K=32) WMMA steps over the current chunk ----
        #pragma unroll
        for (int ks = 0; ks < 2; ++ks) {
            // A fragment: 16-bit A 16x32 layout -> two ds_load_b128 per lane
            ABFrag a;
            const unsigned short* ab = &cur[arow * KCHUNK + ks * 32 + half * 8];
            a.q[0] = *(const uint4*)ab;          // K = half*8 .. half*8+7
            a.q[1] = *(const uint4*)(ab + 16);   // K = 16+half*8 .. 16+half*8+7

            #pragma unroll
            for (int j = 0; j < 5; ++j) {
                // B fragment: lane holds N = j*16+l15, K = half*8 .. half*8+7, 16+half*8 .. 16+half*8+7
                ABFrag b;
                const unsigned short* wb =
                    Wcomb + (size_t)(j * 16 + l15) * HDIM + c * KCHUNK + ks * 32 + half * 8;
                b.q[0] = *(const uint4*)wb;
                b.q[1] = *(const uint4*)(wb + 16);

                acc[j] = __builtin_amdgcn_wmma_f32_16x16x32_bf16(
                    false, a.v, false, b.v, (short)0, acc[j], false, false);
            }
        }
        // One barrier per chunk: guarantees (a) next buffer fully staged before
        // its reads, (b) current buffer's WMMA reads done before it is restaged.
        __syncthreads();
    }

    // ---- spill accumulators: C/D layout value(M,N) @ vgpr=M&7, lane=(M>>3)*16+N ----
    #pragma unroll
    for (int j = 0; j < 5; ++j)
        #pragma unroll
        for (int r = 0; r < 8; ++r)
            ldsE[(wave * 16 + half * 8 + r) * EPI_STRIDE + j * 16 + l15] = acc[j][r];
    __syncthreads();

    // ---- per-token epilogue ----
    if (tid < TOKPB) {
        const float* rowp = &ldsE[tid * EPI_STRIDE];
        float s = b_e2[0];
        #pragma unroll 8
        for (int n = 0; n < CDIM; ++n) {
            float h = rowp[n] + ldsB1[n];
            h = fmaxf(h, 0.0f);
            s = fmaf(h, ldsW2[n], s);
        }
        const size_t g = (size_t)tokBase + tid;
        out[g]             = sigmoidf(rowp[CDIM] + c1p[0]);   // semantic_conflict
        out[NTOK + g]      = sigmoidf(s);                     // expectation_violation
        out[2 * NTOK + g]  = 1.0f / 4096.0f;                  // contextual_surprise == 1/S
    }
}

extern "C" void kernel_launch(void* const* d_in, const int* in_sizes, int n_in,
                              void* d_out, int out_size, void* d_ws, size_t ws_size,
                              hipStream_t stream) {
    const float* X    = (const float*)d_in[0];
    // d_in[1] attention_mask: unused by the reference math
    const float* W_ct = (const float*)d_in[2];
    const float* b_ct = (const float*)d_in[3];
    const float* W_cd = (const float*)d_in[4];
    const float* b_cd = (const float*)d_in[5];
    const float* W_e1 = (const float*)d_in[6];
    const float* b_e1 = (const float*)d_in[7];
    const float* W_e2 = (const float*)d_in[8];
    const float* b_e2 = (const float*)d_in[9];
    // d_in[10..13] (W_q,b_q,W_k,b_k): eliminated — mean(softmax) over same axis == 1/S

    unsigned short* Wcomb = (unsigned short*)d_ws;                       // 80*768 bf16
    float* c1 = (float*)((char*)d_ws + (size_t)NCOLS * HDIM * 2);        // 1 float

    prep_kernel<<<3, 256, 0, stream>>>(W_ct, W_cd, b_ct, b_cd, W_e1, Wcomb, c1);
    incongruity_kernel<<<NTOK / TOKPB, 256, 0, stream>>>(
        X, Wcomb, c1, b_e1, W_e2, b_e2, (float*)d_out);
}
